// FastWaveformer_72602127171906
// MI455X (gfx1250) — matrix operand order repaired and checked by reading, not compiled, not run on hardware
//
#include <hip/hip_runtime.h>
#include <math.h>

// ---------------- model dims ----------------
#define LNUM 8
#define WWIN 32
#define HNUM 8
#define DHEAD 64
#define DMOD 512     // H*DH
#define DFF 2048
#define CCLS 256
#define MMEL 80
#define TSTEPS 512

typedef __bf16 bf16;
typedef __attribute__((ext_vector_type(16))) __bf16 v16bf;
typedef __attribute__((ext_vector_type(8)))  float  v8f;
typedef __attribute__((ext_vector_type(4)))  unsigned v4u;

union VConv { v4u q[2]; v16bf v; };

// ---------------- helpers ----------------
__device__ inline float gelu_tanh(float x) {
  float x3 = x * x * x;
  return 0.5f * x * (1.0f + tanhf(0.7978845608028654f * (x + 0.044715f * x3)));
}

// Load one 32-wide K chunk of the A (weight) and B (broadcast activation)
// operands in ISA 16-bit WMMA layout.
#define LOADAB(Av, Bv, K0)                              \
  do {                                                  \
    (Av).q[0] = *(const v4u*)(arow + (K0));             \
    (Av).q[1] = *(const v4u*)(arow + (K0) + 16);        \
    (Bv).q[0] = *(const v4u*)(xp + (K0));               \
    (Bv).q[1] = *(const v4u*)(xp + (K0) + 16);          \
  } while (0)

// GEMV row-tile (16 rows x KT) as WMMA, 4-deep software pipeline so ~4 K-chunks
// of L2 traffic are in flight behind each v_wmma.  Result: lane 0 holds rows
// r0..r0+7 in acc[0..7], lane 16 holds rows r0+8..r0+15.
template <int KT>
__device__ inline v8f wmma_rowtile(const bf16* __restrict__ Wt,   // W + r0*KT (row-major)
                                   const bf16* __restrict__ x,    // KT activations
                                   int lane) {
  static_assert(KT % 128 == 0, "KT must be multiple of 128");
  const int m  = lane & 15;
  const int hh = lane >> 4;                        // half-wave select
  const bf16* arow = Wt + (size_t)m * KT + hh * 8; // ISA 16-bit A layout
  const bf16* xp   = x + hh * 8;                   // ISA 16-bit B layout (bcast cols)
  v8f acc = {0.f, 0.f, 0.f, 0.f, 0.f, 0.f, 0.f, 0.f};

  VConv a0, b0, a1, b1, a2, b2, a3, b3;
  LOADAB(a0, b0, 0);
  LOADAB(a1, b1, 32);
  LOADAB(a2, b2, 64);
  LOADAB(a3, b3, 96);

  int k = 128;
  for (int c = 0; c < KT / 128 - 1; ++c, k += 128) {
    __builtin_prefetch(arow + k + 256, 0, 3);  // WGP-scope streaming hint
    acc = __builtin_amdgcn_wmma_f32_16x16x32_bf16(false, a0.v, false, b0.v,
                                                  (short)0, acc, false, false);
    LOADAB(a0, b0, k);
    acc = __builtin_amdgcn_wmma_f32_16x16x32_bf16(false, a1.v, false, b1.v,
                                                  (short)0, acc, false, false);
    LOADAB(a1, b1, k + 32);
    acc = __builtin_amdgcn_wmma_f32_16x16x32_bf16(false, a2.v, false, b2.v,
                                                  (short)0, acc, false, false);
    LOADAB(a2, b2, k + 64);
    acc = __builtin_amdgcn_wmma_f32_16x16x32_bf16(false, a3.v, false, b3.v,
                                                  (short)0, acc, false, false);
    LOADAB(a3, b3, k + 96);
  }
  // drain
  acc = __builtin_amdgcn_wmma_f32_16x16x32_bf16(false, a0.v, false, b0.v,
                                                (short)0, acc, false, false);
  acc = __builtin_amdgcn_wmma_f32_16x16x32_bf16(false, a1.v, false, b1.v,
                                                (short)0, acc, false, false);
  acc = __builtin_amdgcn_wmma_f32_16x16x32_bf16(false, a2.v, false, b2.v,
                                                (short)0, acc, false, false);
  acc = __builtin_amdgcn_wmma_f32_16x16x32_bf16(false, a3.v, false, b3.v,
                                                (short)0, acc, false, false);
  return acc;
}

// device-wide barrier among persistent workgroups
__device__ inline void grid_sync(unsigned* cnt, unsigned* gen, unsigned nwg) {
  __syncthreads();
  if (threadIdx.x == 0) {
    __threadfence();
    volatile unsigned* vgen = gen;
    unsigned g = *vgen;
    if (atomicAdd(cnt, 1u) == nwg - 1) {
      atomicExch(cnt, 0u);
      __threadfence();
      atomicAdd(gen, 1u);
    } else {
      while (*vgen == g) { __builtin_amdgcn_s_sleep(2); }
    }
    __threadfence();
  }
  __syncthreads();
}

// block LayerNorm over 512 f32 in LDS -> bf16 in LDS (256 threads)
__device__ inline void block_ln(const float* __restrict__ xin_s,
                                const float* __restrict__ w,
                                const float* __restrict__ b,
                                bf16* __restrict__ out_s, float* red) {
  const int tid = threadIdx.x, lane = tid & 31, wv = tid >> 5;
  float a0 = xin_s[tid], a1 = xin_s[tid + 256];
  float s = a0 + a1, s2 = a0 * a0 + a1 * a1;
#pragma unroll
  for (int m = 16; m >= 1; m >>= 1) {
    s += __shfl_xor(s, m, 32);
    s2 += __shfl_xor(s2, m, 32);
  }
  if (lane == 0) { red[wv] = s; red[8 + wv] = s2; }
  __syncthreads();
  if (tid == 0) {
    float S = 0.f, S2 = 0.f;
#pragma unroll
    for (int i = 0; i < 8; ++i) { S += red[i]; S2 += red[8 + i]; }
    float mu = S * (1.0f / DMOD);
    float var = S2 * (1.0f / DMOD) - mu * mu;
    red[0] = mu; red[1] = rsqrtf(var + 1e-5f);
  }
  __syncthreads();
  float mu = red[0], rs = red[1];
  out_s[tid]       = (bf16)((a0 - mu) * rs * w[tid] + b[tid]);
  out_s[tid + 256] = (bf16)((a1 - mu) * rs * w[tid + 256] + b[tid + 256]);
  __syncthreads();
}

// ---------------- prep kernels ----------------
__global__ void k_cvt_bf16(const float* __restrict__ s, bf16* __restrict__ d, int n) {
  for (int i = blockIdx.x * blockDim.x + threadIdx.x; i < n; i += gridDim.x * blockDim.x)
    d[i] = (bf16)s[i];
}
__global__ void k_tanh(const float* __restrict__ s, float* __restrict__ d, int n) {
  for (int i = blockIdx.x * blockDim.x + threadIdx.x; i < n; i += gridDim.x * blockDim.x)
    d[i] = tanhf(s[i]);
}
__global__ void k_ycond(const float* __restrict__ cw, const float* __restrict__ y,
                        float* __restrict__ yc) {
  int i = blockIdx.x * blockDim.x + threadIdx.x;  // T*D threads
  int t = i >> 9, d = i & 511;
  float s = 0.f;
#pragma unroll
  for (int m = 0; m < MMEL; ++m) s += cw[d * MMEL + m] * y[m * TSTEPS + t];
  yc[(size_t)t * DMOD + d] = s;
}
__global__ void k_init(float* __restrict__ xcin, const float* __restrict__ embt,
                       unsigned* __restrict__ bar) {
  int i = threadIdx.x;
  xcin[i]       = embt[(CCLS / 2 - 1) * DMOD + i];
  xcin[i + 256] = embt[(CCLS / 2 - 1) * DMOD + i + 256];
  if (i == 0) { bar[0] = 0u; bar[1] = 0u; }
}

// ---------------- persistent decode ----------------
struct DecArgs {
  const float *ycond, *samples, *embt;
  const float *ln_w, *ln_b, *kv_b, *q_b, *o_b, *fln_w, *fln_b, *ff_b1, *ff_b2,
      *head_r, *last_b;
  const bf16 *wkv, *wq, *wo, *w1, *w2, *wlast;
  float *xcin, *xcb, *hq, *logits;
  bf16 *h, *cache;
  int* out;
  unsigned* bar;
};

__launch_bounds__(256, 2)
__global__ void FastWaveformer_decode(DecArgs A) {
  __shared__ __align__(16) float xc2_s[DMOD];
  __shared__ __align__(16) float xcb_s[DMOD];
  __shared__ __align__(16) float qs[DMOD];
  __shared__ __align__(16) float attnW[HNUM * WWIN];
  __shared__ __align__(16) float probs[CCLS];
  __shared__ __align__(16) float red[16];
  __shared__ __align__(16) bf16 tmp_bf[DMOD];
  __shared__ __align__(16) bf16 ftmp_bf[DMOD];
  __shared__ __align__(16) bf16 aout[DMOD];
  __shared__ __align__(16) bf16 xl_bf[DMOD];
  __shared__ int s_idx;

  const int tid = threadIdx.x, lane = tid & 31, wv = tid >> 5;
  const int gw = (int)blockIdx.x * 8 + wv;  // global wave id, 0..127
  const unsigned NWG = gridDim.x;

  for (int t = 0; t < TSTEPS; ++t) {
    const float* cond = A.ycond + (size_t)t * DMOD;
    for (int l = 0; l < LNUM; ++l) {
      // ---- Stage A: xc += cond; LN; kv/q GEMVs via WMMA --------------------
      xc2_s[tid]       = A.xcin[tid] + cond[tid];
      xc2_s[tid + 256] = A.xcin[tid + 256] + cond[tid + 256];
      __syncthreads();
      block_ln(xc2_s, A.ln_w + l * DMOD, A.ln_b + l * DMOD, tmp_bf, red);
      if (gw < 64) {  // kv rows 0..1023 -> bf16 into ring cache slot t%32
        int r0 = gw * 16;
        v8f acc = wmma_rowtile<DMOD>(A.wkv + ((size_t)l * 2 * DMOD + r0) * DMOD,
                                     tmp_bf, lane);
        bf16* cslot = A.cache + ((size_t)l * WWIN + (t & 31)) * (2 * DMOD);
        const float* kb = A.kv_b + l * 2 * DMOD;
        if ((lane & 15) == 0) {
          int rb = r0 + (lane >> 4) * 8;
#pragma unroll
          for (int j = 0; j < 8; ++j) cslot[rb + j] = (bf16)(acc[j] + kb[rb + j]);
        }
      } else if (gw < 96) {  // q rows 0..511 -> f32 global
        int r0 = (gw - 64) * 16;
        v8f acc = wmma_rowtile<DMOD>(A.wq + ((size_t)l * DMOD + r0) * DMOD,
                                     tmp_bf, lane);
        const float* qb = A.q_b + l * DMOD;
        if ((lane & 15) == 0) {
          int rb = r0 + (lane >> 4) * 8;
#pragma unroll
          for (int j = 0; j < 8; ++j) A.hq[rb + j] = acc[j] + qb[rb + j];
        }
      }
      grid_sync(A.bar, A.bar + 1, NWG);

      // ---- Stage B: windowed attention (blocks 0..3) + o_w WMMA ------------
      if (blockIdx.x < 4) {
        qs[tid] = A.hq[tid];
        qs[tid + 256] = A.hq[tid + 256];
        __syncthreads();
        {  // one wave per head; lane = window position
          int h = wv, wp = lane;
          int tt = t - (31 - wp);
          float sc = -INFINITY;
          if (tt >= 0) {
            const bf16* kc =
                A.cache + ((size_t)l * WWIN + (tt & 31)) * (2 * DMOD) + h * DHEAD;
            const float* rp =
                A.head_r + ((size_t)l * WWIN + wp) * DMOD + h * DHEAD;
            const float* qp = qs + h * DHEAD;
            float s = 0.f;
#pragma unroll 8
            for (int d = 0; d < DHEAD; ++d) s += ((float)kc[d] + rp[d]) * qp[d];
            sc = s * 0.125f;  // 1/sqrt(64)
          }
          float mx = sc;
#pragma unroll
          for (int m = 16; m >= 1; m >>= 1) mx = fmaxf(mx, __shfl_xor(mx, m, 32));
          float e = (tt >= 0) ? __expf(sc - mx) : 0.f;
          float sm = e;
#pragma unroll
          for (int m = 16; m >= 1; m >>= 1) sm += __shfl_xor(sm, m, 32);
          attnW[h * WWIN + wp] = e / sm;
        }
        __syncthreads();
        {  // out = attn @ v ; lane handles d=lane and d=lane+32
          int h = wv;
          float o0 = 0.f, o1 = 0.f;
          for (int w = 0; w < WWIN; ++w) {
            int tt = t - (31 - w);
            if (tt < 0) continue;
            float aw = attnW[h * WWIN + w];
            const bf16* vc = A.cache + ((size_t)l * WWIN + (tt & 31)) * (2 * DMOD) +
                             DMOD + h * DHEAD;
            o0 += aw * (float)vc[lane];
            o1 += aw * (float)vc[lane + 32];
          }
          aout[h * DHEAD + lane]      = (bf16)o0;
          aout[h * DHEAD + lane + 32] = (bf16)o1;
        }
        __syncthreads();
        {  // o_w GEMV + residual (32 tiles over the 32 waves of blocks 0..3)
          int r0 = gw * 16;
          v8f acc = wmma_rowtile<DMOD>(A.wo + ((size_t)l * DMOD + r0) * DMOD,
                                       aout, lane);
          const float* ob = A.o_b + l * DMOD;
          if ((lane & 15) == 0) {
            int rb = r0 + (lane >> 4) * 8;
#pragma unroll
            for (int j = 0; j < 8; ++j)
              A.xcb[rb + j] = acc[j] + ob[rb + j] + xc2_s[rb + j];
          }
        }
      }
      grid_sync(A.bar, A.bar + 1, NWG);

      // ---- Stage C: LN + ff1 + gelu (128 tiles over all 128 waves) ---------
      xcb_s[tid] = A.xcb[tid];
      xcb_s[tid + 256] = A.xcb[tid + 256];
      __syncthreads();
      block_ln(xcb_s, A.fln_w + l * DMOD, A.fln_b + l * DMOD, ftmp_bf, red);
      {
        int r0 = gw * 16;
        v8f acc = wmma_rowtile<DMOD>(A.w1 + ((size_t)l * DFF + r0) * DMOD,
                                     ftmp_bf, lane);
        const float* b1 = A.ff_b1 + l * DFF;
        if ((lane & 15) == 0) {
          int rb = r0 + (lane >> 4) * 8;
#pragma unroll
          for (int j = 0; j < 8; ++j)
            A.h[rb + j] = (bf16)gelu_tanh(acc[j] + b1[rb + j]);
        }
      }
      grid_sync(A.bar, A.bar + 1, NWG);

      // ---- Stage D: ff2 (K=2048 from global bf16) + residual ---------------
      if (gw < 32) {
        int r0 = gw * 16;
        v8f acc = wmma_rowtile<DFF>(A.w2 + ((size_t)l * DMOD + r0) * DFF,
                                    A.h, lane);
        const float* b2 = A.ff_b2 + l * DMOD;
        if ((lane & 15) == 0) {
          int rb = r0 + (lane >> 4) * 8;
#pragma unroll
          for (int j = 0; j < 8; ++j)
            A.xcin[rb + j] = acc[j] + b2[rb + j] + xcb_s[rb + j];
        }
      }
      grid_sync(A.bar, A.bar + 1, NWG);
    }  // layers

    // ---- Stage E: logits = last_w @ x + last_b (16 tiles, blocks 0..1) -----
    if (blockIdx.x < 2) {
      xl_bf[tid] = (bf16)A.xcin[tid];
      xl_bf[tid + 256] = (bf16)A.xcin[tid + 256];
      __syncthreads();
      if (gw < 16) {
        int r0 = gw * 16;
        v8f acc = wmma_rowtile<DMOD>(A.wlast + (size_t)r0 * DMOD, xl_bf, lane);
        if ((lane & 15) == 0) {
          int rb = r0 + (lane >> 4) * 8;
#pragma unroll
          for (int j = 0; j < 8; ++j) A.logits[rb + j] = acc[j] + A.last_b[rb + j];
        }
      }
    }
    grid_sync(A.bar, A.bar + 1, NWG);

    // ---- Stage F: softmax / cumsum / sample; next x = tanh(emb)[idx] -------
    if (blockIdx.x == 0) {
      float lg = A.logits[tid];
      float mx = lg;
#pragma unroll
      for (int m = 16; m >= 1; m >>= 1) mx = fmaxf(mx, __shfl_xor(mx, m, 32));
      if (lane == 0) red[wv] = mx;
      __syncthreads();
      if (tid == 0) {
        float M = red[0];
#pragma unroll
        for (int i = 1; i < 8; ++i) M = fmaxf(M, red[i]);
        red[0] = M;
      }
      __syncthreads();
      probs[tid] = __expf(lg - red[0]);
      __syncthreads();
      if (tid == 0) {
        float tot = 0.f;
        for (int i = 0; i < CCLS; ++i) tot += probs[i];
        float u = A.samples[t], accm = 0.f;
        int idx = -1;
        for (int i = 0; i < CCLS; ++i) {
          accm += probs[i];
          if (idx < 0 && accm > u * tot) idx = i;
        }
        if (idx < 0) idx = 0;
        s_idx = idx;
        A.out[t] = idx;
      }
      __syncthreads();
      int idx = s_idx;
      A.xcin[tid]       = A.embt[(size_t)idx * DMOD + tid];
      A.xcin[tid + 256] = A.embt[(size_t)idx * DMOD + tid + 256];
    }
    grid_sync(A.bar, A.bar + 1, NWG);
  }  // t
}

// ---------------- host launcher ----------------
extern "C" void kernel_launch(void* const* d_in, const int* in_sizes, int n_in,
                              void* d_out, int out_size, void* d_ws, size_t ws_size,
                              hipStream_t stream) {
  (void)in_sizes; (void)n_in; (void)out_size; (void)ws_size;
  const float* y      = (const float*)d_in[0];
  const float* samp   = (const float*)d_in[1];
  const float* emb    = (const float*)d_in[2];
  const float* condW  = (const float*)d_in[3];
  const float* ln_w   = (const float*)d_in[4];
  const float* ln_b   = (const float*)d_in[5];
  const float* kv_w   = (const float*)d_in[6];
  const float* kv_b   = (const float*)d_in[7];
  const float* q_w    = (const float*)d_in[8];
  const float* q_b    = (const float*)d_in[9];
  const float* o_w    = (const float*)d_in[10];
  const float* o_b    = (const float*)d_in[11];
  const float* fln_w  = (const float*)d_in[12];
  const float* fln_b  = (const float*)d_in[13];
  const float* ff_w1  = (const float*)d_in[14];
  const float* ff_b1  = (const float*)d_in[15];
  const float* ff_w2  = (const float*)d_in[16];
  const float* ff_b2  = (const float*)d_in[17];
  const float* head_r = (const float*)d_in[18];
  const float* last_w = (const float*)d_in[19];
  const float* last_b = (const float*)d_in[20];

  size_t off = 0;
  auto carve = [&](size_t bytes) -> void* {
    void* p = (char*)d_ws + off;
    off += (bytes + 255) & ~(size_t)255;
    return p;
  };
  float* embt   = (float*)carve((size_t)CCLS * DMOD * 4);
  float* ycond  = (float*)carve((size_t)TSTEPS * DMOD * 4);
  bf16* wkv     = (bf16*)carve((size_t)LNUM * 2 * DMOD * DMOD * 2);
  bf16* wq      = (bf16*)carve((size_t)LNUM * DMOD * DMOD * 2);
  bf16* wo      = (bf16*)carve((size_t)LNUM * DMOD * DMOD * 2);
  bf16* w1      = (bf16*)carve((size_t)LNUM * DFF * DMOD * 2);
  bf16* w2      = (bf16*)carve((size_t)LNUM * DMOD * DFF * 2);
  bf16* wlast   = (bf16*)carve((size_t)CCLS * DMOD * 2);
  bf16* cache   = (bf16*)carve((size_t)LNUM * WWIN * 2 * DMOD * 2);
  float* xcin   = (float*)carve(DMOD * 4);
  float* xcb    = (float*)carve(DMOD * 4);
  float* hq     = (float*)carve(DMOD * 4);
  float* logits = (float*)carve(CCLS * 4);
  bf16* hbuf    = (bf16*)carve(DFF * 2);
  unsigned* bar = (unsigned*)carve(2 * 4);

  // prep: bf16 weights, tanh(emb), ycond, init state + barrier
  k_tanh<<<512, 256, 0, stream>>>(emb, embt, CCLS * DMOD);
  k_ycond<<<(TSTEPS * DMOD) / 256, 256, 0, stream>>>(condW, y, ycond);
  k_cvt_bf16<<<4096, 256, 0, stream>>>(kv_w, wkv, LNUM * 2 * DMOD * DMOD);
  k_cvt_bf16<<<4096, 256, 0, stream>>>(q_w, wq, LNUM * DMOD * DMOD);
  k_cvt_bf16<<<4096, 256, 0, stream>>>(o_w, wo, LNUM * DMOD * DMOD);
  k_cvt_bf16<<<4096, 256, 0, stream>>>(ff_w1, w1, LNUM * DFF * DMOD);
  k_cvt_bf16<<<4096, 256, 0, stream>>>(ff_w2, w2, LNUM * DMOD * DFF);
  k_cvt_bf16<<<512, 256, 0, stream>>>(last_w, wlast, CCLS * DMOD);
  k_init<<<1, 256, 0, stream>>>(xcin, embt, bar);

  DecArgs A;
  A.ycond = ycond; A.samples = samp; A.embt = embt;
  A.ln_w = ln_w; A.ln_b = ln_b; A.kv_b = kv_b; A.q_b = q_b; A.o_b = o_b;
  A.fln_w = fln_w; A.fln_b = fln_b; A.ff_b1 = ff_b1; A.ff_b2 = ff_b2;
  A.head_r = head_r; A.last_b = last_b;
  A.wkv = wkv; A.wq = wq; A.wo = wo; A.w1 = w1; A.w2 = w2; A.wlast = wlast;
  A.xcin = xcin; A.xcb = xcb; A.hq = hq; A.logits = logits;
  A.h = hbuf; A.cache = cache;
  A.out = (int*)d_out;
  A.bar = bar;

  FastWaveformer_decode<<<16, 256, 0, stream>>>(A);
}
